// localLSTM_slow_21620865368072
// MI455X (gfx1250) — matrix operand order repaired and checked by reading, not compiled, not run on hardware
//
#include <hip/hip_runtime.h>
#include <hip/hip_bf16.h>

// ---------------------------------------------------------------------------
// Persistent fused LSTM for MI455X (gfx1250, wave32, WMMA bf16).
//   NT=365, NGRID=2048, NX=16, NH=256, 4H=1024, NY=1
// One workgroup (256 thr = 8 waves) owns MB=64 grid rows for all timesteps.
// Per step: gates = concat(relu(x@Win^T+b), h) [64x512] @ Wpack [512x1024]
// via v_wmma_f32_16x16x32_bf16; h/c state stays in LDS; x staged by TDM.
// ---------------------------------------------------------------------------

#define NT    365
#define NGRID 2048
#define NX    16
#define NH    256
#define NG    1024          // 4*NH
#define MB    64            // grid rows per workgroup
#define NWG   (NGRID / MB)  // 32 workgroups

// padded LDS strides (avoid 64-bank conflicts)
#define ASTR 520            // A buffer row stride in bf16 (512 + 8)
#define GSTR 1032           // gates row stride in bf16 (1024 + 8)
#define CSTR 260            // c row stride in f32

typedef __attribute__((ext_vector_type(16))) __bf16 v16bf;
typedef __attribute__((ext_vector_type(8)))  float  v8f;
typedef __attribute__((ext_vector_type(4)))  float  f32x4;
typedef unsigned int u32x4 __attribute__((ext_vector_type(4)));
typedef int          i32x8 __attribute__((ext_vector_type(8)));
typedef int          i32x4 __attribute__((ext_vector_type(4)));

union Frag16 { v16bf v; f32x4 f[2]; };

// LDS layout offsets (bytes)
#define OFF_A     0
#define OFF_G     (OFF_A + MB * ASTR * 2)            //  66560
#define OFF_C     (OFF_G + MB * GSTR * 2)            // +132096
#define OFF_XST   (OFF_C + MB * CSTR * 4)            // +66560
#define OFF_XA    (OFF_XST + 2 * MB * NX * 4)        // +8192
#define OFF_BIN   (OFF_XA + MB * 32 * 2)             // +4096
#define OFF_BG    (OFF_BIN + NH * 4)                 // +1024
#define OFF_WOUT  (OFF_BG + NG * 4)                  // +4096
#define SMEM_SZ   (OFF_WOUT + NH * 4)                // total ~283648 B

// ---------------------------------------------------------------------------
// Weight pre-pack kernels: convert f32 row-major weights into bf16 WMMA
// B-fragment order: frag[lane][j] with n = nt*16 + (lane&15),
// k = kt*32 + (lane>>4)*16 + j  -> each lane loads 2x contiguous b128.
// ---------------------------------------------------------------------------
__global__ void pack_win(const float* __restrict__ w_in, __bf16* __restrict__ dst) {
    int idx = blockIdx.x * 256 + threadIdx.x;     // 16 ntiles * 32 lanes * 16
    if (idx >= 16 * 32 * 16) return;
    int j  = idx & 15;
    int l  = (idx >> 4) & 31;
    int nt = idx >> 9;
    int n  = nt * 16 + (l & 15);
    int k  = ((l >> 4) << 4) + j;                 // 0..31 (K padded to 32)
    float v = (k < NX) ? w_in[n * NX + k] : 0.0f;
    dst[idx] = (__bf16)v;
}

__global__ void pack_wg(const float* __restrict__ w_ih, const float* __restrict__ w_hh,
                        __bf16* __restrict__ dst) {
    int idx = blockIdx.x * 256 + threadIdx.x;     // 64 nt * 16 kt * 32 * 16
    int j  = idx & 15;
    int l  = (idx >> 4) & 31;
    int kt = (idx >> 9) & 15;
    int nt = idx >> 13;
    int n  = nt * 16 + (l & 15);                  // gate index 0..1023
    int k  = kt * 32 + ((l >> 4) << 4) + j;       // 0..511 (x0 | h)
    float v = (k < NH) ? w_ih[n * NH + k] : w_hh[n * NH + (k - NH)];
    dst[idx] = (__bf16)v;
}

// ---------------------------------------------------------------------------
// TDM: 1-D contiguous tensor load (nelem f32) global -> LDS.  D# group0/1 per
// CDNA5 ISA; groups 2/3 unused (<=2D). This toolchain uses
// the 6-arg builtin form: (u32x4, i32x8, i32x4, i32x4, i32x8, cpol).
// ---------------------------------------------------------------------------
__device__ __forceinline__ void tdm_load_1d_f32(const float* gsrc, unsigned int lds_off,
                                                int nelem) {
    unsigned long long ga = (unsigned long long)gsrc;
    u32x4 g0;
    g0[0] = 1u;                                            // count=1, user desc
    g0[1] = lds_off;                                       // lds_addr
    g0[2] = (unsigned int)ga;                              // global_addr[31:0]
    g0[3] = (unsigned int)((ga >> 32) & 0x1FFFFFFULL) | (2u << 30); // addr hi | type=2
    i32x8 g1;
    g1[0] = (2 << 16);                                     // data_size = 4B
    g1[1] = (nelem & 0xFFFF) << 16;                        // tensor_dim0 lo16
    g1[2] = (nelem >> 16) & 0xFFFF;                        // tensor_dim0 hi16
    g1[3] = (nelem & 0xFFFF) << 16;                        // tile_dim0
    g1[4] = 0;                                             // tile_dim1/2 unused
    g1[5] = nelem;                                         // tensor_dim0_stride
    g1[6] = 0;
    g1[7] = 0;
    i32x4 z4 = {0, 0, 0, 0};
    i32x8 z8 = {0, 0, 0, 0, 0, 0, 0, 0};
    __builtin_amdgcn_tensor_load_to_lds(g0, g1, z4, z4, z8, 0);
}

__device__ __forceinline__ float fast_sigmoid(float x) {
    return 1.0f / (1.0f + __expf(-x));
}
__device__ __forceinline__ float fast_tanh(float x) {
    float e = __expf(-2.0f * x);
    return (1.0f - e) / (1.0f + e);
}

// ---------------------------------------------------------------------------
// Main persistent kernel.
// ---------------------------------------------------------------------------
__global__ __launch_bounds__(256, 1)
void lstm_persistent(const float* __restrict__ x,
                     const float* __restrict__ b_in,
                     const float* __restrict__ b_ih,
                     const float* __restrict__ b_hh,
                     const float* __restrict__ w_out,
                     const float* __restrict__ b_out,
                     const __bf16* __restrict__ WinP,   // [16][32][16]
                     const __bf16* __restrict__ WgP,    // [64][16][32][16]
                     float* __restrict__ out) {
    extern __shared__ __align__(16) char smem[];
    __bf16* Abuf   = (__bf16*)(smem + OFF_A);     // [MB][ASTR]: cols 0-255 x0, 256-511 h
    __bf16* Gbuf   = (__bf16*)(smem + OFF_G);     // [MB][GSTR] raw gates
    float*  Cst    = (float*) (smem + OFF_C);     // [MB][CSTR] cell state
    float*  Xst    = (float*) (smem + OFF_XST);   // 2 x [MB][NX] TDM staging
    __bf16* Xa     = (__bf16*)(smem + OFF_XA);    // [MB][32] bf16, K-padded
    float*  BiasIn = (float*) (smem + OFF_BIN);   // [NH]
    float*  BiasG  = (float*) (smem + OFF_BG);    // [NG] = b_ih + b_hh
    float*  Wout   = (float*) (smem + OFF_WOUT);  // [NH]

    const int tid  = threadIdx.x;
    const int wave = tid >> 5;
    const int lane = tid & 31;
    const int row0 = blockIdx.x * MB;

    // ---- one-time init: biases, output weights, zero h and c -------------
    for (int i = tid; i < NH; i += 256) { BiasIn[i] = b_in[i]; Wout[i] = w_out[i]; }
    for (int i = tid; i < NG; i += 256) { BiasG[i] = b_ih[i] + b_hh[i]; }
    for (int i = tid; i < MB * NH; i += 256) {
        int r = i >> 8, c = i & 255;
        Cst[r * CSTR + c] = 0.0f;
        Abuf[r * ASTR + 256 + c] = (__bf16)0.0f;
    }
    // prime TDM pipeline: x block for t=0
    if (wave == 0) {
        tdm_load_1d_f32(x + (size_t)row0 * NX,
                        (unsigned int)(size_t)(void*)(Xst), MB * NX);
    }
    __syncthreads();

    const float bout = b_out[0];
    const int mtile  = wave & 3;          // 4 M-tiles of 16 rows
    const int arow   = mtile * 16 + (lane & 15);
    const int kb     = (lane >> 4) << 4;  // lane-group K base (0 or 16)
    const int nhalf  = wave >> 2;         // gate-column half (0..1)

    #pragma unroll 1
    for (int t = 0; t < NT; ++t) {
        // ---- wait for TDM x-block, convert f32 -> bf16 K-padded ----------
        if (wave == 0) __builtin_amdgcn_s_wait_tensorcnt(0);
        __syncthreads();
        {
            const float* xs = Xst + (t & 1) * MB * NX;
            for (int i = tid; i < MB * 32; i += 256) {
                int r = i >> 5, k = i & 31;
                Xa[i] = (__bf16)((k < NX) ? xs[r * NX + k] : 0.0f);
            }
        }
        __syncthreads();
        // overlap next step's x fetch with this step's GEMMs
        if (wave == 0 && (t + 1) < NT) {
            tdm_load_1d_f32(x + ((size_t)(t + 1) * NGRID + row0) * NX,
                            (unsigned int)(size_t)(void*)(Xst + ((t + 1) & 1) * MB * NX),
                            MB * NX);
        }

        // ---- GEMM1: x0 = relu(Xa[64x32] @ WinP -> [64x256]) --------------
        {
            Frag16 a;
            const __bf16* ap = Xa + arow * 32 + kb;
            a.f[0] = *(const f32x4*)(ap);        // K = kb..kb+7
            a.f[1] = *(const f32x4*)(ap + 8);    // K = kb+8..kb+15
            const int rbase = mtile * 16 + ((lane >> 4) << 3);
            #pragma unroll
            for (int i = 0; i < 8; ++i) {
                int nt = nhalf * 8 + i;
                Frag16 b;
                const __bf16* bp = WinP + ((size_t)nt * 32 + lane) * 16;
                b.f[0] = *(const f32x4*)(bp);
                b.f[1] = *(const f32x4*)(bp + 8);
                v8f acc = {};
                acc = __builtin_amdgcn_wmma_f32_16x16x32_bf16(
                        false, a.v, false, b.v, (short)0, acc, false, false);
                int col = nt * 16 + (lane & 15);
                float bia = BiasIn[col];
                #pragma unroll
                for (int r = 0; r < 8; ++r) {
                    float v = acc[r] + bia;
                    v = v > 0.0f ? v : 0.0f;
                    Abuf[(rbase + r) * ASTR + col] = (__bf16)v;
                }
            }
        }
        __syncthreads();

        // ---- GEMM2: gates = Abuf[64x512] @ WgP -> [64x1024] --------------
        {
            const int rbase = mtile * 16 + ((lane >> 4) << 3);
            #pragma unroll 1
            for (int p = 0; p < 4; ++p) {
                int nbase = nhalf * 32 + p * 8;      // 8 N-tiles per pass
                v8f acc[8];
                #pragma unroll
                for (int i = 0; i < 8; ++i) acc[i] = (v8f){};
                #pragma unroll 1
                for (int kt = 0; kt < 16; ++kt) {
                    Frag16 a;
                    const __bf16* ap = Abuf + arow * ASTR + kt * 32 + kb;
                    a.f[0] = *(const f32x4*)(ap);
                    a.f[1] = *(const f32x4*)(ap + 8);
                    const __bf16* bp = WgP + (((size_t)nbase * 16 + kt) * 32 + lane) * 16;
                    #pragma unroll
                    for (int i = 0; i < 8; ++i) {
                        Frag16 b;
                        const __bf16* bpi = bp + (size_t)i * 16 * 32 * 16;
                        b.f[0] = *(const f32x4*)(bpi);
                        b.f[1] = *(const f32x4*)(bpi + 8);
                        acc[i] = __builtin_amdgcn_wmma_f32_16x16x32_bf16(
                                   false, a.v, false, b.v, (short)0, acc[i], false, false);
                    }
                }
                #pragma unroll
                for (int i = 0; i < 8; ++i) {
                    int col = (nbase + i) * 16 + (lane & 15);
                    #pragma unroll
                    for (int r = 0; r < 8; ++r)
                        Gbuf[(rbase + r) * GSTR + col] = (__bf16)acc[i][r];
                }
            }
        }
        __syncthreads();

        // ---- elementwise LSTM cell + h write-back + out projection -------
        {
            const int row = tid >> 2;       // 64 rows
            const int q   = tid & 3;        // 4 lanes per row (same wave)
            const __bf16* g   = Gbuf + row * GSTR;
            float*       crow = Cst  + row * CSTR;
            __bf16*      hrow = Abuf + row * ASTR + 256;
            float oacc = 0.0f;
            #pragma unroll 4
            for (int jj = 0; jj < 64; ++jj) {
                int j = q * 64 + jj;
                float gi = (float)g[j]        + BiasG[j];
                float gf = (float)g[256 + j]  + BiasG[256 + j];
                float gg = (float)g[512 + j]  + BiasG[512 + j];
                float go = (float)g[768 + j]  + BiasG[768 + j];
                float i_ = fast_sigmoid(gi);
                float f_ = fast_sigmoid(gf);
                float g_ = fast_tanh(gg);
                float o_ = fast_sigmoid(go);
                float c  = f_ * crow[j] + i_ * g_;
                crow[j]  = c;
                float h  = o_ * fast_tanh(c);
                hrow[j]  = (__bf16)h;
                oacc    += h * Wout[j];
            }
            oacc += __shfl_xor(oacc, 1);
            oacc += __shfl_xor(oacc, 2);
            if (q == 0) out[(size_t)t * NGRID + row0 + row] = oacc + bout;
        }
        __syncthreads();
    }
}

// ---------------------------------------------------------------------------
extern "C" void kernel_launch(void* const* d_in, const int* in_sizes, int n_in,
                              void* d_out, int out_size, void* d_ws, size_t ws_size,
                              hipStream_t stream) {
    const float* x     = (const float*)d_in[0];
    const float* w_in  = (const float*)d_in[1];
    const float* b_in  = (const float*)d_in[2];
    const float* w_ih  = (const float*)d_in[3];
    const float* w_hh  = (const float*)d_in[4];
    const float* b_ih  = (const float*)d_in[5];
    const float* b_hh  = (const float*)d_in[6];
    const float* w_out = (const float*)d_in[7];
    const float* b_out = (const float*)d_in[8];
    float* out = (float*)d_out;

    __bf16* WinP = (__bf16*)d_ws;                 // 8192 elems (16 KB)
    __bf16* WgP  = WinP + 16384;                  // 524288 elems (1 MB)

    (void)hipFuncSetAttribute((const void*)lstm_persistent,
                              hipFuncAttributeMaxDynamicSharedMemorySize, SMEM_SZ);

    pack_win<<<32, 256, 0, stream>>>(w_in, WinP);
    pack_wg<<<2048, 256, 0, stream>>>(w_ih, w_hh, WgP);
    lstm_persistent<<<NWG, 256, SMEM_SZ, stream>>>(
        x, b_in, b_ih, b_hh, w_out, b_out, WinP, WgP, out);
}
